// ControlModel_g_33990371181239
// MI455X (gfx1250) — matrix operand order repaired and checked by reading, not compiled, not run on hardware
//
#include <hip/hip_runtime.h>
#include <hip/hip_bf16.h>

typedef __attribute__((ext_vector_type(16))) _Float16 v16h;
typedef __attribute__((ext_vector_type(8)))  _Float16 v8h;
typedef __attribute__((ext_vector_type(8)))  float    v8f;

#define LL    16
#define NSITE 256
#define NH    128
#define NB    32
#define NTR   16

// ---------------------------------------------------------------------------
// Kernel 1: swizzle W1 (f32, 256x128) into f16 hi/lo arrays in exact WMMA
// B-fragment order: element index = ((c*8 + nt)*32 + lane)*16 + slot
// where k = c*32 + (slot<8 ? (lane>>4)*8 + slot : 16 + (lane>>4)*8 + slot-8), h = nt*16 + (lane&15).
// Per lane a fragment is then 32 contiguous bytes (two aligned b128 loads).
// ---------------------------------------------------------------------------
__global__ void prep_w1(const float* __restrict__ W1,
                        _Float16* __restrict__ whi,
                        _Float16* __restrict__ wlo) {
  int tid  = blockIdx.x * blockDim.x + threadIdx.x;   // 0..32767
  int slot = tid & 15;
  int lane = (tid >> 4) & 31;
  int nt   = (tid >> 9) & 7;
  int cc   = tid >> 12;
  int k = cc * 32 + ((slot < 8) ? ((lane >> 4) * 8 + slot) : (16 + (lane >> 4) * 8 + (slot - 8)));
  int h = nt * 16 + (lane & 15);
  float v = W1[k * NH + h];
  _Float16 hi = (_Float16)v;
  _Float16 lo = (_Float16)(v - (float)hi);
  whi[tid] = hi;
  wlo[tid] = lo;
}

// ---------------------------------------------------------------------------
// Kernel 2: one workgroup per (transform, batch). 512 threads = 16 waves.
// Each wave owns one 16-row M-tile of the 256x128 = (shifts x hidden) GEMM,
// K = 256 lattice sites, done as 8 K-chunks of v_wmma_f32_16x16x32_f16 with
// a hi/lo f16 split (3 WMMAs) for ~f32 accuracy. Epilogue: tanh + weighted
// reduction. Deterministic: per-WG partial written to Fpart[wg], no atomics.
// ---------------------------------------------------------------------------
__global__ __launch_bounds__(512) void fno_kernel(
    const float* __restrict__ x,  const float* __restrict__ W1,
    const float* __restrict__ b1, const float* __restrict__ W2,
    const float* __restrict__ b2, const _Float16* __restrict__ whi,
    const _Float16* __restrict__ wlo, float* __restrict__ Fpart) {
  __shared__ float    s_tx[LL][LL];      // transformed lattice (f32)
  __shared__ float    s_fx[NSITE];       // force field
  __shared__ _Float16 s_txh[LL][32];     // lattice hi, columns duplicated
  __shared__ _Float16 s_txl[LL][32];     // lattice lo, columns duplicated
  __shared__ float    s_b1[NH], s_w2[NH], s_c[NH];
  __shared__ float    s_red[16];

  const int e  = threadIdx.x;            // 0..511
  const int wg = blockIdx.x;             // 0..511
  const int b  = wg & 31;                // batch
  const int t  = wg >> 5;                // transform 0..15

  // ---- apply symmetry transform: sign * flip^f( rot90^k( x[b] ) ) ----
  if (e < NSITE) {
    int i = e >> 4, j = e & 15;
    float sgn = (t < 8) ? 1.f : -1.f;
    int tt = t & 7, rk = tt >> 1, flip = tt & 1;
    int i2 = flip ? (15 - i) : i;
    int j2 = j;
    int si, sj;
    switch (rk) {                         // rot90(m,k): out[i,j] = in[map(i,j)]
      case 0:  si = i2;      sj = j2;      break;
      case 1:  si = j2;      sj = 15 - i2; break;
      case 2:  si = 15 - i2; sj = 15 - j2; break;
      default: si = 15 - j2; sj = i2;      break;
    }
    float v = sgn * x[b * NSITE + si * LL + sj];
    s_tx[i][j] = v;
    _Float16 hi = (_Float16)v;
    _Float16 lo = (_Float16)(v - (float)hi);
    s_txh[i][j] = hi; s_txh[i][j + 16] = hi;   // duplicate for wrap-free reads
    s_txl[i][j] = lo; s_txl[i][j + 16] = lo;
  } else if (e < NSITE + NH) {
    int h = e - NSITE;
    float w2 = W2[h];
    s_b1[h] = b1[h];
    s_w2[h] = w2;
    s_c[h]  = w2 * W1[h];                 // c[h] = W2[h] * W1[0][h]
  }
  __syncthreads();

  float partial = 0.f;
  const float b2v = b2[0];

  // ---- force: fx = 2k*nbr - 2*phi - 4*lam*phi*(phi^2-1), k=.25 lam=.5 ----
  if (e < NSITE) {
    int i = e >> 4, j = e & 15;
    float p   = s_tx[i][j];
    float nbr = s_tx[(i + 15) & 15][j] + s_tx[(i + 1) & 15][j] +
                s_tx[i][(j + 15) & 15] + s_tx[i][(j + 1) & 15];
    float f = 0.5f * nbr - 2.f * p - 2.f * p * (p * p - 1.f);
    s_fx[e] = f;
    partial += b2v * f;                   // b2 * sum_s fx[s] term
  }
  __syncthreads();

  // ---- GEMM: pre[s,h] = sum_k S[s,k] * W1[k,h] via WMMA f16 split ----
  const int lane = e & 31;
  const int wv   = e >> 5;                // M-tile (0..15)
  const int hf   = lane >> 4;             // half of wave
  const int nn   = lane & 15;
  const int srow = wv * 16 + nn;          // A-matrix row = shift index s
  const int dy   = srow >> 4, dx = srow & 15;
  const int cb   = dx + 8 * hf;           // column base into duplicated rows

  v8f acc[8] = {};
  for (int c = 0; c < 8; ++c) {           // K-chunk: sites rows 2c, 2c+1
    int r0 = (dy + 2 * c)     & 15;
    int r1 = (dy + 2 * c + 1) & 15;
    v16h ah, al;
#pragma unroll
    for (int q = 0; q < 8; ++q) {         // ISA 16-bit A 16x32 layout
      ah[q]     = s_txh[r0][cb + q];
      ah[q + 8] = s_txh[r1][cb + q];
      al[q]     = s_txl[r0][cb + q];
      al[q + 8] = s_txl[r1][cb + q];
    }
#pragma unroll
    for (int nt = 0; nt < 8; ++nt) {      // N-tile of 16 hidden units
      int base = ((c * 8 + nt) * 32 + lane) * 16;
      const v8h* ph = (const v8h*)(whi + base);
      const v8h* pl = (const v8h*)(wlo + base);
      v8h bh0 = ph[0], bh1 = ph[1];
      v8h bl0 = pl[0], bl1 = pl[1];
      v16h bh, bl;
#pragma unroll
      for (int q = 0; q < 8; ++q) {
        bh[q] = bh0[q]; bh[q + 8] = bh1[q];
        bl[q] = bl0[q]; bl[q + 8] = bl1[q];
      }
      acc[nt] = __builtin_amdgcn_wmma_f32_16x16x32_f16(
          false, ah, false, bh, (short)0, acc[nt], false, false);
      acc[nt] = __builtin_amdgcn_wmma_f32_16x16x32_f16(
          false, ah, false, bl, (short)0, acc[nt], false, false);
      acc[nt] = __builtin_amdgcn_wmma_f32_16x16x32_f16(
          false, al, false, bh, (short)0, acc[nt], false, false);
    }
  }

  // ---- epilogue: t=tanh(pre+b1); add c[h](1-t^2) + fx[s]*W2[h]*t ----
#pragma unroll
  for (int nt = 0; nt < 8; ++nt) {
    int h = nt * 16 + nn;                 // C layout: col = lane&15
    float w2 = s_w2[h], cc = s_c[h], bb = s_b1[h];
#pragma unroll
    for (int v = 0; v < 8; ++v) {         // C layout: row = v + 8*hf
      int sl = wv * 16 + v + 8 * hf;
      float th = tanhf(acc[nt][v] + bb);
      partial += cc * (1.f - th * th) + s_fx[sl] * w2 * th;
    }
  }

  // ---- reduce: wave shuffle, then LDS across 16 waves ----
#pragma unroll
  for (int off = 16; off > 0; off >>= 1)
    partial += __shfl_xor(partial, off, 32);
  if (lane == 0) s_red[wv] = partial;
  __syncthreads();
  if (e == 0) {
    float tot = 0.f;
    for (int w = 0; w < 16; ++w) tot += s_red[w];
    Fpart[wg] = tot;                      // deterministic, no atomics
  }
}

// ---------------------------------------------------------------------------
// Kernel 3: correlator O[b], sum transform partials, final scalar loss.
// ---------------------------------------------------------------------------
__global__ __launch_bounds__(512) void finish_kernel(
    const float* __restrict__ x, const float* __restrict__ muO,
    const float* __restrict__ Fpart, float* __restrict__ out) {
  __shared__ float s_x0[NB][LL];
  __shared__ float s_m[LL];
  __shared__ float s_v[NB];
  int e = threadIdx.x;
  {
    int b = e >> 4, j = e & 15;           // 512 threads cover 32x16
    float s = 0.f;
    for (int i = 0; i < LL; ++i) s += x[b * NSITE + i * LL + j];
    s_x0[b][j] = s * (1.f / 16.f);
  }
  __syncthreads();
  if (e < LL) {
    float s = 0.f;
    for (int b = 0; b < NB; ++b) s += s_x0[b][e];
    s_m[e] = s * (1.f / 32.f);
  }
  __syncthreads();
  if (e < NB) {
    float O = 0.f;
    for (int j = 0; j < LL; ++j) {
      int j2 = (j + 4) & 15;
      O += (s_x0[e][j] - s_m[j]) * (s_x0[e][j2] - s_m[j2]);
    }
    O *= (1.f / 16.f);
    float F = 0.f;
    for (int tt = 0; tt < NTR; ++tt) F += Fpart[tt * NB + e];
    F *= (1.f / 16.f);                    // mean over transforms
    float d = (O - F) - muO[0];
    s_v[e] = d * d;
  }
  __syncthreads();
  if (e == 0) {
    float s = 0.f;
    for (int b = 0; b < NB; ++b) s += s_v[b];
    out[0] = s * (1.f / 32.f);
  }
}

// ---------------------------------------------------------------------------
extern "C" void kernel_launch(void* const* d_in, const int* in_sizes, int n_in,
                              void* d_out, int out_size, void* d_ws, size_t ws_size,
                              hipStream_t stream) {
  const float* x   = (const float*)d_in[0];   // (32,16,16)
  const float* W1  = (const float*)d_in[1];   // (256,128)
  const float* b1  = (const float*)d_in[2];   // (128,)
  const float* W2  = (const float*)d_in[3];   // (128,)
  const float* b2  = (const float*)d_in[4];   // (1,)
  const float* muO = (const float*)d_in[5];   // (1,)

  _Float16* whi   = (_Float16*)d_ws;                       // 32768 f16 = 64KB
  _Float16* wlo   = whi + 32768;                           // 64KB
  float*    Fpart = (float*)((char*)d_ws + 131072);        // 512 f32

  prep_w1<<<128, 256, 0, stream>>>(W1, whi, wlo);
  fno_kernel<<<512, 512, 0, stream>>>(x, W1, b1, W2, b2, whi, wlo, Fpart);
  finish_kernel<<<1, 512, 0, stream>>>(x, muO, Fpart, (float*)d_out);
}
